// SelfAttention_74655121539564
// MI455X (gfx1250) — matrix operand order repaired and checked by reading, not compiled, not run on hardware
//
#include <hip/hip_runtime.h>
#include <hip/hip_bf16.h>

// ---------------------------------------------------------------------------
// Types for WMMA fragments (gfx1250, wave32)
// ---------------------------------------------------------------------------
typedef __attribute__((ext_vector_type(16))) __bf16       v16bf;
typedef __attribute__((ext_vector_type(8)))  float        v8f;
typedef __attribute__((ext_vector_type(4)))  float        f32x4;
typedef __attribute__((ext_vector_type(4)))  unsigned int u32x4;
typedef __attribute__((ext_vector_type(4)))  int          i32x4;

union BF16x16 {
    v16bf          v;
    unsigned short s[16];
    u32x4          q[2];
};

__device__ __forceinline__ unsigned short f2bf(float f) {
    unsigned u = __builtin_bit_cast(unsigned, f);
    u += 0x7FFFu + ((u >> 16) & 1u);          // round-to-nearest-even
    return (unsigned short)(u >> 16);
}

#define WMMA_BF16(A, B, C) \
    __builtin_amdgcn_wmma_f32_16x16x32_bf16(false, (A), false, (B), (short)0, (C), false, false)

// ---------------------------------------------------------------------------
// Async global->LDS copies (CDNA5 ASYNCcnt path), with sync fallback
// ---------------------------------------------------------------------------
#if defined(__has_builtin)
#if __has_builtin(__builtin_amdgcn_global_load_async_to_lds_b128) && \
    __has_builtin(__builtin_amdgcn_s_wait_asynccnt)
#define USE_ASYNC_LDS 1
#endif
#endif

#define AS1 __attribute__((address_space(1)))
#define AS3 __attribute__((address_space(3)))
typedef AS3 unsigned short lds_us;

__device__ __forceinline__ void async_copy16(const void* g, lds_us* l) {
#ifdef USE_ASYNC_LDS
    __builtin_amdgcn_global_load_async_to_lds_b128((AS1 i32x4*)(g), (AS3 i32x4*)(l), 0, 0);
#else
    *(AS3 u32x4*)l = *(const u32x4*)g;
#endif
}

__device__ __forceinline__ void async_commit() {
#ifdef USE_ASYNC_LDS
    __builtin_amdgcn_s_wait_asynccnt(0);
#endif
}

// Problem constants
#define NTOK   16384      // 4 * 4096 tokens
#define SEQ    4096
#define EMB    768
#define DHEAD  64
#define NOUT   192        // 3 * 64

// ---------------------------------------------------------------------------
// Kernel 0: convert W (f32 [192,768]) to bf16 (L2-resident, reused 1024x)
// ---------------------------------------------------------------------------
__global__ void cvt_w_kernel(const float* __restrict__ W, unsigned short* __restrict__ Wb, int n) {
    int i = blockIdx.x * blockDim.x + threadIdx.x;
    if (i < n) Wb[i] = f2bf(W[i]);
}

// ---------------------------------------------------------------------------
// Kernel 1: fused KQV projection with WMMA bf16.
//   8 waves / block share a double-buffered 12KB W k-slab in LDS (async
//   staged, branchless wrapped index, pointer-swapped buffers).  B fragments
//   are hand double-buffered so fragment t+1 loads while WMMA t executes.
// ---------------------------------------------------------------------------
__global__ void __launch_bounds__(256, 1)
proj_kernel(const float* __restrict__ z,
            const unsigned short* __restrict__ Wb,
            const float* __restrict__ bias,
            unsigned short* __restrict__ Kb,
            unsigned short* __restrict__ Qb,
            unsigned short* __restrict__ Vt) {
    __shared__ __attribute__((aligned(16))) unsigned short Wt[2][NOUT][32]; // 24KB

    const int tid  = threadIdx.x;          // 0..255
    const int lane = tid & 31;
    const int wave = tid >> 5;
    const int m0   = (blockIdx.x * 8 + wave) * 16;
    const int col  = lane & 15;
    const int half = lane >> 4;

    v8f acc[12] = {};
    const float* zp = z + (size_t)(m0 + col) * EMB;

    // Staging: slab = 192 rows x 32 K bf16 = 768 x 16B chunks; 3 per thread.
    const int c0 = tid, c1 = 256 + tid, c2 = 512 + tid;
    const unsigned short* g0 = Wb + (size_t)(c0 >> 2) * EMB + (c0 & 3) * 8;
    const unsigned short* g1 = Wb + (size_t)(c1 >> 2) * EMB + (c1 & 3) * 8;
    const unsigned short* g2 = Wb + (size_t)(c2 >> 2) * EMB + (c2 & 3) * 8;
    lds_us* l0 = (lds_us*)&Wt[0][c0 >> 2][(c0 & 3) * 8];
    lds_us* l1 = (lds_us*)&Wt[0][c1 >> 2][(c1 & 3) * 8];
    lds_us* l2 = (lds_us*)&Wt[0][c2 >> 2][(c2 & 3) * 8];
    const int BUFO = NOUT * 32;            // elements between the two buffers

    // ---- prologue: stage k-slab 0 into buffer 0 --------------------------
    async_copy16(g0, l0);
    async_copy16(g1, l1);
    async_copy16(g2, l2);
    async_commit();

    unsigned short (*wcur)[32] = &Wt[0][0];
    int nxt = BUFO;                        // element offset of the "next" buffer

#pragma clang loop unroll(disable)
    for (int k0 = 0; k0 < EMB; k0 += 32) {
        __syncthreads();   // current slab published; next buffer free

        // ---- branchless: stage slab (k0+32, wrapped) into next buffer ----
        const int kn = (k0 + 32 == EMB) ? 0 : (k0 + 32);   // uniform select
        async_copy16(g0 + kn, l0 + nxt);
        async_copy16(g1 + kn, l1 + nxt);
        async_copy16(g2 + kn, l2 + nxt);

        // ---- A fragment from z (f32 -> bf16) ----
        f32x4 a[4];
        a[0] = *(const f32x4*)(zp + k0 + half * 8);
        a[1] = *(const f32x4*)(zp + k0 + half * 8 + 4);
        a[2] = *(const f32x4*)(zp + k0 + 16 + half * 8);
        a[3] = *(const f32x4*)(zp + k0 + 16 + half * 8 + 4);
        BF16x16 A;
#pragma unroll
        for (int i = 0; i < 16; ++i) A.s[i] = f2bf(a[i >> 2][i & 3]);

        // ---- 12 WMMAs fed from LDS, hand double-buffered B fragments ----
        BF16x16 Bf[2];
        {
            const unsigned short* wp = &wcur[col][half * 8];
            Bf[0].q[0] = *(const u32x4*)(wp);
            Bf[0].q[1] = *(const u32x4*)(wp + 16);
        }
#pragma unroll
        for (int t = 0; t < 12; ++t) {
            if (t + 1 < 12) {
                const unsigned short* wp = &wcur[(t + 1) * 16 + col][half * 8];
                Bf[(t + 1) & 1].q[0] = *(const u32x4*)(wp);
                Bf[(t + 1) & 1].q[1] = *(const u32x4*)(wp + 16);
            }
            acc[t] = WMMA_BF16(A.v, Bf[t & 1].v, acc[t]);
        }

        async_commit();                    // staged copies durable pre-barrier
        wcur = (unsigned short (*)[32])((unsigned short*)wcur + nxt);
        nxt  = -nxt;                       // swap buffers
    }

    // Epilogue: bias + de-interleave.  D layout: lane, vgpr r -> row r+8*half, col lane%16
#pragma unroll
    for (int t = 0; t < 12; ++t) {
        const int   o   = t * 16 + col;
        const float bv  = bias[o];
        const int   mat = o % 3;
        const int   d   = o / 3;
#pragma unroll
        for (int r = 0; r < 8; ++r) {
            const int tok = m0 + r + 8 * half;
            const unsigned short h = f2bf(acc[t][r] + bv);
            if (mat == 0)      Kb[(size_t)tok * DHEAD + d] = h;
            else if (mat == 1) Qb[(size_t)tok * DHEAD + d] = h;
            else               Vt[(size_t)d * NTOK + tok]  = h;   // transposed V
        }
    }
}

// ---------------------------------------------------------------------------
// Kernel 2: flash-style attention.
//   8 waves / block share double-buffered K/V tiles in LDS staged with async
//   global->LDS copies; B fragments preloaded ahead of each WMMA burst.
// ---------------------------------------------------------------------------
__global__ void __launch_bounds__(256, 1)
attn_kernel(const unsigned short* __restrict__ Kb,
            const unsigned short* __restrict__ Qb,
            const unsigned short* __restrict__ Vt,
            float* __restrict__ out) {
    __shared__ __attribute__((aligned(16))) unsigned short Kt [2][32][64]; // [key][dim]
    __shared__ __attribute__((aligned(16))) unsigned short Vtl[2][64][32]; // [dim][key]
    __shared__ __attribute__((aligned(16))) unsigned short Plds[8][16][32];

    const int tid   = threadIdx.x;
    const int lane  = tid & 31;
    const int wave  = tid >> 5;
    const int unit  = blockIdx.x * 8 + wave;     // 0..1023
    const int batch = unit >> 8;
    const int qtile = unit & 255;
    const int base  = batch << 12;               // token offset of batch
    const int q0    = qtile << 4;
    const int col   = lane & 15;
    const int half  = lane >> 4;

    // Cooperative-copy bases (16B chunks, 256 threads)
    const int kkey = tid >> 3, ksub = tid & 7;   // K tile: 32 keys x 8 chunks
    const int vdim = tid >> 2, vsub = tid & 3;   // V tile: 64 dims x 4 chunks
    const unsigned short* kg = Kb + (size_t)(base + kkey) * DHEAD + ksub * 8;
    const unsigned short* vg = Vt + (size_t)vdim * NTOK + base + vsub * 8;
    lds_us* kl = (lds_us*)&Kt [0][kkey][ksub * 8];
    lds_us* vl = (lds_us*)&Vtl[0][vdim][vsub * 8];
    const int KBUFO = 32 * 64;                   // elements between K buffers
    const int VBUFO = 64 * 32;                   // elements between V buffers

    // Q A-fragments for dim chunks [0,32) and [32,64)
    const unsigned short* qp = Qb + (size_t)(base + q0 + col) * DHEAD;
    BF16x16 Aq[2];
#pragma unroll
    for (int h = 0; h < 2; ++h) {
        const int d0 = h * 32;
        Aq[h].q[0] = *(const u32x4*)(qp + d0 + half * 8);
        Aq[h].q[1] = *(const u32x4*)(qp + d0 + 16 + half * 8);
    }

    float mrow[8], lrow[8];
#pragma unroll
    for (int r = 0; r < 8; ++r) { mrow[r] = -1e30f; lrow[r] = 0.0f; }
    v8f o_acc[4] = {};

    // ---- prologue: stage tile j0=0 into buffer 0 -------------------------
    async_copy16(kg, kl);
    async_copy16(vg, vl);
    async_commit();

    unsigned short (*kcur)[64] = &Kt [0][0];
    unsigned short (*vcur)[32] = &Vtl[0][0];
    int knxt = KBUFO, vnxt = VBUFO;

#pragma clang loop unroll(disable)
    for (int j0 = 0; j0 < SEQ; j0 += 32) {
        __syncthreads();   // current tile published; next buffer free

        // ---- branchless: stage tile (j0+32, wrapped) into next buffer ----
        const int jn = (j0 + 32) & (SEQ - 1);
        async_copy16(kg + (size_t)jn * DHEAD, kl + knxt);
        async_copy16(vg + jn,                 vl + vnxt);

        // ---- S = Q @ K^T for 32 keys: preload 4 B fragments, then burst --
        BF16x16 Bk[4];
#pragma unroll
        for (int jt = 0; jt < 2; ++jt)
#pragma unroll
            for (int h = 0; h < 2; ++h) {
                const unsigned short* kp = &kcur[jt * 16 + col][h * 32 + half * 8];
                Bk[jt * 2 + h].q[0] = *(const u32x4*)(kp);
                Bk[jt * 2 + h].q[1] = *(const u32x4*)(kp + 16);
            }
        v8f s[2] = {};
#pragma unroll
        for (int jt = 0; jt < 2; ++jt)
#pragma unroll
            for (int h = 0; h < 2; ++h)
                s[jt] = WMMA_BF16(Aq[h].v, Bk[jt * 2 + h].v, s[jt]);

        // ---- online softmax over the 32 new columns ----------------------
        float corr[8], p0[8], p1[8];
#pragma unroll
        for (int r = 0; r < 8; ++r) {
            float s0 = s[0][r] * 0.125f;    // 1/sqrt(64)
            float s1 = s[1][r] * 0.125f;
            float mx = fmaxf(s0, s1);
#pragma unroll
            for (int msk = 1; msk <= 8; msk <<= 1)       // reduce within 16-lane half
                mx = fmaxf(mx, __shfl_xor(mx, msk, 32));
            const float mnew = fmaxf(mrow[r], mx);
            corr[r] = __expf(mrow[r] - mnew);
            p0[r]   = __expf(s0 - mnew);
            p1[r]   = __expf(s1 - mnew);
            float sum = p0[r] + p1[r];
#pragma unroll
            for (int msk = 1; msk <= 8; msk <<= 1)
                sum += __shfl_xor(sum, msk, 32);
            lrow[r] = lrow[r] * corr[r] + sum;
            mrow[r] = mnew;
        }

        // ---- P: C-layout -> A-layout through per-wave LDS ----------------
#pragma unroll
        for (int r = 0; r < 8; ++r) {
            Plds[wave][r + 8 * half][col]      = f2bf(p0[r]);
            Plds[wave][r + 8 * half][col + 16] = f2bf(p1[r]);
        }
        BF16x16 Ap;
        const unsigned short* pp = &Plds[wave][col][0];
        Ap.q[0] = *(const u32x4*)(pp + half * 8);
        Ap.q[1] = *(const u32x4*)(pp + 16 + half * 8);

        // ---- O = O*corr + P @ V: preload 4 B fragments, then burst -------
        BF16x16 Bv[4];
#pragma unroll
        for (int t = 0; t < 4; ++t) {
            const unsigned short* vp = &vcur[t * 16 + col][half * 8];
            Bv[t].q[0] = *(const u32x4*)(vp);
            Bv[t].q[1] = *(const u32x4*)(vp + 16);
        }
#pragma unroll
        for (int t = 0; t < 4; ++t) {
#pragma unroll
            for (int r = 0; r < 8; ++r) o_acc[t][r] *= corr[r];
            o_acc[t] = WMMA_BF16(Ap.v, Bv[t].v, o_acc[t]);
        }

        async_commit();                    // staged copies durable pre-barrier
        kcur = (unsigned short (*)[64])((unsigned short*)kcur + knxt);
        vcur = (unsigned short (*)[32])((unsigned short*)vcur + vnxt);
        knxt = -knxt;
        vnxt = -vnxt;
    }

    // ---- normalize and store --------------------------------------------
#pragma unroll
    for (int t = 0; t < 4; ++t) {
#pragma unroll
        for (int r = 0; r < 8; ++r) {
            const int tok = base + q0 + r + 8 * half;
            out[(size_t)tok * DHEAD + t * 16 + col] = o_acc[t][r] / lrow[r];
        }
    }
}

// ---------------------------------------------------------------------------
// Launcher
// ---------------------------------------------------------------------------
extern "C" void kernel_launch(void* const* d_in, const int* in_sizes, int n_in,
                              void* d_out, int out_size, void* d_ws, size_t ws_size,
                              hipStream_t stream) {
    (void)in_sizes; (void)n_in; (void)out_size; (void)ws_size;

    const float* z    = (const float*)d_in[0];   // [4,4096,768]
    const float* W    = (const float*)d_in[1];   // [192,768]
    const float* bias = (const float*)d_in[2];   // [192]
    float* out        = (float*)d_out;           // [4,4096,64]

    char* ws = (char*)d_ws;
    const size_t WB_BYTES = (size_t)NOUT * EMB * 2;          // 294912
    const size_t KQ_BYTES = (size_t)NTOK * DHEAD * 2;        // 2 MiB each
    unsigned short* Wb = (unsigned short*)(ws);
    unsigned short* Kb = (unsigned short*)(ws + WB_BYTES);
    unsigned short* Qb = (unsigned short*)(ws + WB_BYTES + KQ_BYTES);
    unsigned short* Vt = (unsigned short*)(ws + WB_BYTES + 2 * KQ_BYTES);

    cvt_w_kernel<<<(NOUT * EMB + 255) / 256, 256, 0, stream>>>(W, Wb, NOUT * EMB);
    proj_kernel<<<NTOK / 16 / 8, 256, 0, stream>>>(z, Wb, bias, Kb, Qb, Vt);
    attn_kernel<<<NTOK / 16 / 8, 256, 0, stream>>>(Kb, Qb, Vt, out);
}
